// Attention_76656576299268
// MI455X (gfx1250) — compile-verified, matrix operand order checked by reading, not run on hardware
//
#include <hip/hip_runtime.h>
#include <hip/hip_bf16.h>
#include <math.h>

// ---------------- problem constants ----------------
constexpr int B_   = 8;
constexpr int S_   = 32;
constexpr int D_   = 4096;
constexpr int H_   = 32;
constexpr int KVH_ = 8;
constexpr int HD_  = 128;
constexpr int SW_  = 4096;
constexpr int PREV_ = SW_ - S_;   // 4064
constexpr int REP_ = H_ / KVH_;   // 4
constexpr int T_   = SW_;         // total kv length
#define SCALE_F 0.08838834764831845f  // 1/sqrt(128)

// ---------------- WMMA types / helpers ----------------
typedef __attribute__((ext_vector_type(16))) __bf16 v16bf;
typedef __attribute__((ext_vector_type(8)))  float  v8f;

__device__ inline v8f wmma_bf16(v16bf a, v16bf b, v8f c) {
  // D = A(16x32 bf16) * B(32x16 bf16) + C(16x16 f32)
  return __builtin_amdgcn_wmma_f32_16x16x32_bf16(
      /*neg_a=*/false, a, /*neg_b=*/false, b,
      /*c_mod=*/(short)0, c, /*reuse_a=*/false, /*reuse_b=*/false);
}

// Load a 16x32 bf16 fragment (A-matrix layout, ISA 7.12.2) from LDS.
// p0 points at element (row0, k0); stride in bf16 elements between rows.
// Lane (lane&15) supplies the row; lane>=16 takes the +8 K-offset half.
// The B fragment (32x16, column n per lane) uses the identical gather when
// the B matrix is stored N-major (row n holds the 32 K values).
__device__ inline v16bf lds_frag(const __bf16* p0, int stride, int lane) {
  const int row = lane & 15;
  const int kg  = (lane >> 4) << 3;   // 0 or 8
  const __bf16* p = p0 + row * stride + kg;
  v16bf f;
#pragma unroll
  for (int i = 0; i < 4; ++i) {
    f[2 * i + 0]     = p[2 * i + 0];        // VGPR i   : K = 2i,2i+1 (+kg)
    f[2 * i + 1]     = p[2 * i + 1];
    f[8 + 2 * i + 0] = p[16 + 2 * i + 0];   // VGPR 4+i : K = 16+2i   (+kg)
    f[8 + 2 * i + 1] = p[16 + 2 * i + 1];
  }
  return f;
}

// Pack two f32 into adjacent bf16 (one aligned 32-bit LDS store).
__device__ inline void pack2_bf16(__bf16* dst, float lo, float hi) {
  union { __bf16 h[2]; unsigned u; } pk;
  pk.h[0] = (__bf16)lo;
  pk.h[1] = (__bf16)hi;
  *(unsigned*)dst = pk.u;
}

// =====================================================================
// Generic GEMM: C[M,N] = A[M,K] * W[K,N], f32 in memory, bf16 WMMA math.
// BM=32, BN=256, BK=32. 256 threads = 8 waves; each wave computes a
// 16x64 slab (4 adjacent 16x16 N-tiles) -> 4 WMMA per A-fragment.
// =====================================================================
__global__ __launch_bounds__(256) void gemm_bf16_wmma(
    const float* __restrict__ A, const float* __restrict__ W,
    float* __restrict__ C, int M, int N, int K) {
  __shared__ __bf16 As[32][36];    // padded rows
  __shared__ __bf16 Bs[256][36];   // W tile transposed: Bs[n][k]

  const int tid  = threadIdx.x;
  const int lane = tid & 31;
  const int wave = tid >> 5;
  const int mT   = wave & 1;       // M tile (0..1)
  const int nQ   = wave >> 1;      // 64-col slab (0..3)
  const int bM   = blockIdx.y * 32;
  const int bN   = blockIdx.x * 256;

  v8f acc[4] = {};

  for (int k0 = 0; k0 < K; k0 += 32) {
    // ---- A tile 32x32: 256 float4, one per thread, contiguous stores ----
    {
      const int row = tid >> 3;
      const int c4  = (tid & 7) << 2;
      const float4 a4 =
          *(const float4*)(A + (size_t)(bM + row) * K + k0 + c4);
      pack2_bf16(&As[row][c4 + 0], a4.x, a4.y);
      pack2_bf16(&As[row][c4 + 2], a4.z, a4.w);
    }
    // ---- W tile 32x256 transposed, k-pair packed b32 stores ----
    // unit: (k-pair kp, 4 n's) -> 2 float4 loads, 4 b32 LDS stores
#pragma unroll
    for (int j = 0; j < 4; ++j) {
      const int idx = tid + j * 256;       // 1024 units
      const int kp  = idx >> 6;            // 0..15
      const int n4  = (idx & 63) << 2;     // 0..252
      const float* s0 = W + (size_t)(k0 + 2 * kp) * N + bN + n4;
      const float4 w0 = *(const float4*)s0;
      const float4 w1 = *(const float4*)(s0 + N);
      pack2_bf16(&Bs[n4 + 0][2 * kp], w0.x, w1.x);
      pack2_bf16(&Bs[n4 + 1][2 * kp], w0.y, w1.y);
      pack2_bf16(&Bs[n4 + 2][2 * kp], w0.z, w1.z);
      pack2_bf16(&Bs[n4 + 3][2 * kp], w0.w, w1.w);
      if (k0 + 32 < K)  // pull next K-slice of W toward L2
        __builtin_prefetch(s0 + (size_t)32 * N, 0, 1);
    }
    __syncthreads();

    const v16bf a = lds_frag(&As[mT * 16][0], 36, lane);
#pragma unroll
    for (int t = 0; t < 4; ++t) {
      const v16bf b = lds_frag(&Bs[(nQ * 4 + t) * 16][0], 36, lane);
      acc[t] = wmma_bf16(a, b, acc[t]);
    }
    __syncthreads();
  }

  // ---- store C (f32 C/D layout: row = 8*(lane>=16)+r, col = lane&15) ----
  const int rbase = (lane >> 4) * 8;
  const int col   = lane & 15;
#pragma unroll
  for (int t = 0; t < 4; ++t) {
#pragma unroll
    for (int r = 0; r < 8; ++r) {
      const int grow = bM + mT * 16 + rbase + r;
      C[(size_t)grow * N + bN + (nQ * 4 + t) * 16 + col] = acc[t][r];
    }
  }
}

// =====================================================================
// RoPE on q (B,S,H,HD) and new k (B,S,KVH,HD), interleaved pairs.
// freqs_{cos,sin}: (1,S,HD/2)
// =====================================================================
__global__ __launch_bounds__(256) void rope_kernel(
    float* __restrict__ q, float* __restrict__ k,
    const float* __restrict__ fcos, const float* __restrict__ fsin) {
  const int idx  = blockIdx.x * blockDim.x + threadIdx.x;
  const int p    = idx & 63;                  // pair index 0..63
  const int rest = idx >> 6;
  const int head = rest % (H_ + KVH_);
  const int bs   = rest / (H_ + KVH_);        // b*S + s, 0..255
  const int s    = bs & (S_ - 1);
  const float c  = fcos[s * (HD_ / 2) + p];
  const float sn = fsin[s * (HD_ / 2) + p];
  float* t;
  if (head < H_)
    t = q + ((size_t)bs * H_ + head) * HD_ + 2 * p;
  else
    t = k + ((size_t)bs * KVH_ + (head - H_)) * HD_ + 2 * p;
  const float tr = t[0], ti = t[1];
  t[0] = tr * c - ti * sn;
  t[1] = tr * sn + ti * c;
}

// =====================================================================
// Flash attention over T=4096 in chunks of 32. One workgroup per (b,h).
// Waves 0..3 build the 32x32 score chunk (WMMA), wave 0 runs the online
// softmax, all 8 waves do P*V (WMMA) into their 16x32 O slab.
// =====================================================================
__global__ __launch_bounds__(256) void attn_kernel(
    const float* __restrict__ q,        // (B,S,H,HD) roped
    const float* __restrict__ xk,       // (B,S,KVH,HD) roped
    const float* __restrict__ xv,       // (B,S,KVH,HD)
    const float* __restrict__ cache_k,  // (SW,KVH,HD)
    const float* __restrict__ cache_v,  // (SW,KVH,HD)
    const float* __restrict__ mask,     // (B,S,S)
    float* __restrict__ out)            // (B,S,H*HD)
{
  __shared__ __bf16 Qs[32][136];   // Q tile, pre-scaled by 1/sqrt(HD)
  __shared__ __bf16 Ks[32][136];   // K chunk rows (B frag of Q*K^T as-is)
  __shared__ __bf16 Vt[128][36];   // V chunk transposed: Vt[d][t]
  __shared__ __bf16 Pb[32][36];    // softmaxed probabilities (bf16)
  __shared__ float  Ps[32][32];    // raw score chunk
  __shared__ float  mrun[32], lrun[32], cfac[32];

  const int tid  = threadIdx.x;
  const int lane = tid & 31;
  const int wave = tid >> 5;
  const int bh = blockIdx.x;
  const int b  = bh / H_;
  const int h  = bh % H_;
  const int kh = h / REP_;

  // ---- load Q, scale by 1/sqrt(HD): 1024 float4, 4 per thread ----
#pragma unroll
  for (int j = 0; j < 4; ++j) {
    const int idx = tid + j * 256;
    const int row = idx >> 5;
    const int c4  = (idx & 31) << 2;
    const float4 v =
        *(const float4*)(q + (((size_t)(b * S_ + row)) * H_ + h) * HD_ + c4);
    pack2_bf16(&Qs[row][c4 + 0], v.x * SCALE_F, v.y * SCALE_F);
    pack2_bf16(&Qs[row][c4 + 2], v.z * SCALE_F, v.w * SCALE_F);
  }
  if (tid < 32) { mrun[tid] = -INFINITY; lrun[tid] = 0.f; }

  v8f acc0 = {}, acc1 = {};
  const int mT = wave & 1;   // which 16-row half of the 32 queries
  const int nP = wave >> 1;  // which 32-col slab of HD=128

  for (int c = 0; c < T_ / 32; ++c) {
    const int t0 = c * 32;
    __syncthreads();  // LDS reuse fence (also covers initial Q fill)

    // ---- K chunk rows: 1024 float4, contiguous bf16 stores ----
#pragma unroll
    for (int j = 0; j < 4; ++j) {
      const int idx = tid + j * 256;
      const int r   = idx >> 5;          // t within chunk
      const int c4  = (idx & 31) << 2;   // hd offset
      const int t   = t0 + r;
      const float* sk =
          (t < PREV_)
              ? cache_k + ((size_t)t * KVH_ + kh) * HD_ + c4
              : xk + (((size_t)(b * S_ + (t - PREV_))) * KVH_ + kh) * HD_ + c4;
      const float4 kv = *(const float4*)sk;
      pack2_bf16(&Ks[r][c4 + 0], kv.x, kv.y);
      pack2_bf16(&Ks[r][c4 + 2], kv.z, kv.w);
    }
    // ---- V chunk transposed, t-pair packed b32 stores ----
    // chunk is 32-aligned so a (t, t+1) pair never straddles PREV.
#pragma unroll
    for (int j = 0; j < 2; ++j) {
      const int idx = tid + j * 256;     // 512 units
      const int tp  = idx >> 5;          // t-pair 0..15
      const int c4  = (idx & 31) << 2;   // hd base
      const int ta  = t0 + 2 * tp;
      const float* sv0 =
          (ta < PREV_)
              ? cache_v + ((size_t)ta * KVH_ + kh) * HD_ + c4
              : xv + (((size_t)(b * S_ + (ta - PREV_))) * KVH_ + kh) * HD_ + c4;
      const float4 v0 = *(const float4*)sv0;
      const float4 v1 = *(const float4*)(sv0 + (size_t)KVH_ * HD_);
      pack2_bf16(&Vt[c4 + 0][2 * tp], v0.x, v1.x);
      pack2_bf16(&Vt[c4 + 1][2 * tp], v0.y, v1.y);
      pack2_bf16(&Vt[c4 + 2][2 * tp], v0.z, v1.z);
      pack2_bf16(&Vt[c4 + 3][2 * tp], v0.w, v1.w);
    }
    __syncthreads();

    // ---- scores: S(32x32) = Qs(32x128) * Ks^T, waves 0..3, 4 K-steps ----
    if (wave < 4) {
      const int mi = wave >> 1, ni = wave & 1;
      v8f s = {};
#pragma unroll
      for (int kk = 0; kk < HD_; kk += 32) {
        const v16bf a  = lds_frag(&Qs[mi * 16][kk], 136, lane);
        const v16bf bb = lds_frag(&Ks[ni * 16][kk], 136, lane);
        s = wmma_bf16(a, bb, s);
      }
      const int rbase = (lane >> 4) * 8;
      const int col   = lane & 15;
#pragma unroll
      for (int r = 0; r < 8; ++r) {
        const int row  = mi * 16 + rbase + r;
        const int tcol = t0 + ni * 16 + col;
        float val = s[r];
        if (tcol >= PREV_)
          val += mask[((size_t)b * S_ + row) * S_ + (tcol - PREV_)];
        Ps[row][ni * 16 + col] = val;
      }
    }
    __syncthreads();

    // ---- online softmax, one row per lane of wave 0 ----
    if (tid < 32) {
      const float mold = mrun[tid];
      float mx = mold;
#pragma unroll 8
      for (int j = 0; j < 32; ++j) mx = fmaxf(mx, Ps[tid][j]);
      const float corr = __expf(mold - mx);
      float l = lrun[tid] * corr;
#pragma unroll 4
      for (int j = 0; j < 32; j += 2) {
        const float p0 = __expf(Ps[tid][j + 0] - mx);
        const float p1 = __expf(Ps[tid][j + 1] - mx);
        pack2_bf16(&Pb[tid][j], p0, p1);
        l += p0 + p1;
      }
      mrun[tid] = mx; lrun[tid] = l; cfac[tid] = corr;
    }
    __syncthreads();

    // ---- rescale O, then O += P(16x32) * V-slab, all 8 waves ----
    {
      const int rbase = (lane >> 4) * 8;
#pragma unroll
      for (int r = 0; r < 8; ++r) {
        const float cf = cfac[mT * 16 + rbase + r];
        acc0[r] *= cf;
        acc1[r] *= cf;
      }
      const v16bf a  = lds_frag(&Pb[mT * 16][0], 36, lane);
      const v16bf b0 = lds_frag(&Vt[(nP * 2 + 0) * 16][0], 36, lane);
      const v16bf b1 = lds_frag(&Vt[(nP * 2 + 1) * 16][0], 36, lane);
      acc0 = wmma_bf16(a, b0, acc0);
      acc1 = wmma_bf16(a, b1, acc1);
    }
  }

  // ---- epilogue: divide by l, nan_to_num, store (B,S,H*HD) ----
  {
    const int rbase = (lane >> 4) * 8;
    const int col   = lane & 15;
#pragma unroll
    for (int r = 0; r < 8; ++r) {
      const int row = mT * 16 + rbase + r;
      const float inv = 1.f / lrun[row];
      float o0 = acc0[r] * inv;
      float o1 = acc1[r] * inv;
      if (!__builtin_isfinite(o0)) o0 = 0.f;
      if (!__builtin_isfinite(o1)) o1 = 0.f;
      float* dst = out + ((size_t)(b * S_ + row)) * (H_ * HD_) + h * HD_;
      dst[(nP * 2 + 0) * 16 + col] = o0;
      dst[(nP * 2 + 1) * 16 + col] = o1;
    }
  }
}

// =====================================================================
extern "C" void kernel_launch(void* const* d_in, const int* in_sizes, int n_in,
                              void* d_out, int out_size, void* d_ws,
                              size_t ws_size, hipStream_t stream) {
  const float* x    = (const float*)d_in[0];
  const float* fcos = (const float*)d_in[1];
  const float* fsin = (const float*)d_in[2];
  const float* mask = (const float*)d_in[3];
  const float* ck   = (const float*)d_in[4];
  const float* cv   = (const float*)d_in[5];
  const float* Wq   = (const float*)d_in[6];
  const float* Wk   = (const float*)d_in[7];
  const float* Wv   = (const float*)d_in[8];
  const float* Wo   = (const float*)d_in[9];
  float* out = (float*)d_out;

  float* ws   = (float*)d_ws;
  float* q_ws = ws;                          // 256*4096 f32
  float* k_ws = q_ws + (size_t)256 * 4096;   // 256*1024
  float* v_ws = k_ws + (size_t)256 * 1024;   // 256*1024
  float* a_ws = v_ws + (size_t)256 * 1024;   // 256*4096

  const dim3 blk(256);
  const int M = B_ * S_;  // 256

  // QKV projections (bf16 WMMA)
  gemm_bf16_wmma<<<dim3((H_ * HD_) / 256, M / 32), blk, 0, stream>>>(
      x, Wq, q_ws, M, H_ * HD_, D_);
  gemm_bf16_wmma<<<dim3((KVH_ * HD_) / 256, M / 32), blk, 0, stream>>>(
      x, Wk, k_ws, M, KVH_ * HD_, D_);
  gemm_bf16_wmma<<<dim3((KVH_ * HD_) / 256, M / 32), blk, 0, stream>>>(
      x, Wv, v_ws, M, KVH_ * HD_, D_);

  // RoPE on q and new k
  rope_kernel<<<dim3((M * (H_ + KVH_) * (HD_ / 2)) / 256), blk, 0, stream>>>(
      q_ws, k_ws, fcos, fsin);

  // Flash attention, one workgroup per (b,h)
  attn_kernel<<<dim3(B_ * H_), blk, 0, stream>>>(q_ws, k_ws, v_ws, ck, cv,
                                                 mask, a_ws);

  // Output projection
  gemm_bf16_wmma<<<dim3(D_ / 256, M / 32), blk, 0, stream>>>(
      a_ws, Wo, out, M, D_, H_ * HD_);
}
